// graph_encode_38019050504215
// MI455X (gfx1250) — matrix operand order repaired and checked by reading, not compiled, not run on hardware
//
#include <hip/hip_runtime.h>

typedef __attribute__((ext_vector_type(16))) __bf16 v16bf;
typedef __attribute__((ext_vector_type(8)))  __bf16 v8bf;
typedef __attribute__((ext_vector_type(8)))  float  v8f;
typedef __attribute__((ext_vector_type(4)))  unsigned int v4u;
typedef __attribute__((ext_vector_type(8)))  int v8i;
typedef __attribute__((ext_vector_type(4)))  int v4i;

#define B_  4
#define N_  2048
#define E_  1600
#define H_  512
#define P_  2
#define NH_ 4
#define DH_ 128
#define H4_ 2048
#define M_  (B_*N_)

__device__ __forceinline__ v16bf concat8(v8bf lo, v8bf hi) {
  return __builtin_shufflevector(lo, hi, 0,1,2,3,4,5,6,7,8,9,10,11,12,13,14,15);
}
__device__ __forceinline__ v8f wmma_bf16(v16bf a, v16bf b, v8f c) {
  return __builtin_amdgcn_wmma_f32_16x16x32_bf16(false, a, false, b, (short)0, c, false, false);
}

// ---- Tensor Data Mover: 2-D tile (global, row-major, bf16) -> LDS with row padding ----
// D# per CDNA5 ISA. tensor dims == tile dims (no OOB path).
// pad_interval: code v -> 2^(v+1) DWORDs; pad_amount: code v -> v+1 DWORDs.
// Toolchain here is the 6-arg builtin variant (clang-23 / therock-10.0 headers).
__device__ __forceinline__ void tdm_load_2d(
    const void* lds_dst, const __bf16* gsrc,
    unsigned d0, unsigned d1, unsigned long long stride0,
    unsigned pad_interval, unsigned pad_amount)
{
  unsigned long long ga = (unsigned long long)(uintptr_t)gsrc;
  v4u g0;
  g0[0] = 1u;                                      // count=1, user mode
  g0[1] = (unsigned)(uintptr_t)lds_dst;            // LDS byte offset (aperture low bits)
  g0[2] = (unsigned)ga;                            // global_addr[31:0]
  g0[3] = (unsigned)((ga >> 32) & 0x1FFFFFFull) | (2u << 30);  // addr[56:32], type=2
  unsigned ctrl = (1u << 16)                       // data_size = 1 (2 bytes)
                | (1u << 20)                       // pad_enable
                | (pad_interval << 22)
                | (pad_amount << 25);
  unsigned long long q0 = (unsigned long long)ctrl
                        | ((unsigned long long)(d0 & 0xFFFFu) << 48);      // tensor_dim0[15:0] @48
  unsigned long long q1 = (unsigned long long)(d0 >> 16)                   // tensor_dim0[31:16]
                        | ((unsigned long long)d1 << 16)                   // tensor_dim1 @80
                        | ((unsigned long long)(d0 & 0xFFFFu) << 48);      // tile_dim0 @112
  unsigned long long q2 = (unsigned long long)(d1 & 0xFFFFu)               // tile_dim1 @128
                        | ((stride0 & 0xFFFFFFFFull) << 32);               // dim0_stride[31:0] @160
  unsigned long long q3 = (stride0 >> 32) & 0xFFFFull;                     // dim0_stride[47:32]
  struct Q { unsigned long long a, b, c, d; } qq = {q0, q1, q2, q3};
  v8i g1 = __builtin_bit_cast(v8i, qq);
  v4i gz4 = {0, 0, 0, 0};
  v8i gz8 = {0, 0, 0, 0, 0, 0, 0, 0};
  __builtin_amdgcn_tensor_load_to_lds(g0, g1, gz4, gz4, gz8, 0);
}

// ---------------- cast + transpose weights: W[p][k][n] -> Wt[p][n][k] (bf16) ----------------
__global__ void k_cast_t(const float* __restrict__ src, __bf16* __restrict__ dst, int K, int N) {
  int p = blockIdx.y;
  int i = blockIdx.x * 256 + threadIdx.x;
  if (i >= K * N) return;
  int kk = i / N, n = i % N;
  dst[((size_t)p * N + n) * K + kk] = (__bf16)src[((size_t)p * K + kk) * N + n];
}

// ---------------- build x = concat(vents, renc_w[rels]) ----------------
__global__ void k_build_x(const float* __restrict__ vents, const float* __restrict__ renc,
                          const int* __restrict__ rels,
                          float* __restrict__ xF, __bf16* __restrict__ xB) {
  int i = blockIdx.x * 256 + threadIdx.x;
  if (i >= B_ * N_ * H_) return;
  int col = i % H_;
  int bn  = i / H_;
  int n   = bn % N_;
  int b   = bn / N_;
  float v;
  if (n < E_) v = vents[((size_t)b * E_ + n) * H_ + col];
  else        v = renc[(size_t)rels[b * (N_ - E_) + (n - E_)] * H_ + col];
  xF[i] = v;
  xB[i] = (__bf16)v;
}

// ---------------- WMMA GEMM, TDM double-buffered: out = act(A[MxK] @ W[KxN] + bias) ----------
// Wt is W transposed: [N][K]. 8 waves as 4x2; wave -> 16 rows x 32 cols.
__global__ __launch_bounds__(256) void k_gemm(
    const __bf16* __restrict__ A, const __bf16* __restrict__ Wt,
    const float* __restrict__ bias, const float* __restrict__ prelu,
    float* __restrict__ outF, __bf16* __restrict__ outB,
    int M, int N, int K, float scale, int vtrans)
{
  __shared__ __align__(64) __bf16 sA[2][64][40];    // 64x32 tile, pitch 40 (pad 4 DW / 16 DW)
  __shared__ __align__(64) __bf16 sBt[2][64][48];   // 64x32 tile, pitch 48 (pad 8 DW / 16 DW)

  const int tid  = threadIdx.x;
  const int lane = tid & 31;
  const int wave = tid >> 5;
  const int wRow = wave >> 1, wCol = wave & 1;
  const int half = lane >> 4, l16 = lane & 15;
  const int bm = blockIdx.y * 64, bn = blockIdx.x * 64;
  const bool issuer = (wave == 0);

  v8f acc0 = {}; v8f acc1 = {};

  if (issuer) {
    tdm_load_2d(&sA[0][0][0],  A  + (size_t)bm * K, 32, 64, (unsigned long long)K, 3, 3);
    tdm_load_2d(&sBt[0][0][0], Wt + (size_t)bn * K, 32, 64, (unsigned long long)K, 3, 7);
    __builtin_amdgcn_s_wait_tensorcnt(0);
  }
  __syncthreads();

  const int nk = K >> 5;
  for (int t = 0; t < nk; ++t) {
    const int cur = t & 1;
    if (issuer && (t + 1) < nk) {       // prefetch next tile; buffer free since last barrier
      const int nxt = cur ^ 1;
      tdm_load_2d(&sA[nxt][0][0],  A  + (size_t)bm * K + (t + 1) * 32, 32, 64,
                  (unsigned long long)K, 3, 3);
      tdm_load_2d(&sBt[nxt][0][0], Wt + (size_t)bn * K + (t + 1) * 32, 32, 64,
                  (unsigned long long)K, 3, 7);
    }
    // A fragment: row = l16 (+wave row group); K = half*16 + i (contiguous)
    v8bf alo = *(const v8bf*)&sA[cur][wRow * 16 + l16][half * 16];
    v8bf ahi = *(const v8bf*)&sA[cur][wRow * 16 + l16][half * 16 + 8];
    v16bf afrag = concat8(alo, ahi);
    // B fragments: col = l16, K = half*16 + i (contiguous)
    v16bf bf0 = *(const v16bf*)&sBt[cur][wCol * 32 + l16][half * 16];
    v16bf bf1 = *(const v16bf*)&sBt[cur][wCol * 32 + 16 + l16][half * 16];
    acc0 = wmma_bf16(afrag, bf0, acc0);
    acc1 = wmma_bf16(afrag, bf1, acc1);
    if (issuer && (t + 1) < nk) __builtin_amdgcn_s_wait_tensorcnt(0);
    __syncthreads();
  }

#pragma unroll
  for (int r = 0; r < 8; ++r) {
    int row = bm + wRow * 16 + r + 8 * half;   // C layout: row = r + 8*(lane/16)
#pragma unroll
    for (int t = 0; t < 2; ++t) {
      int col = bn + wCol * 32 + t * 16 + l16;
      float v = (t == 0 ? acc0[r] : acc1[r]) + bias[col];
      v *= scale;
      if (prelu) { float a = prelu[col]; v = v >= 0.f ? v : a * v; }
      size_t o;
      if (vtrans) {                            // write V transposed: [b][h][dh][n]
        int bb = row >> 11, n = row & (N_ - 1);
        int hh = col >> 7,  dh = col & (DH_ - 1);
        o = ((size_t)(bb * NH_ + hh) * DH_ + dh) * N_ + n;
      } else {
        o = (size_t)row * N + col;
      }
      if (outF) outF[o] = v;
      if (outB) outB[o] = (__bf16)v;
    }
  }
}

// ---------------- flash-style masked attention, TDM double-buffered ----------------
// grid: 256 blocks; block = (b, h, 8 q-tiles); 8 waves, one 16-row q-tile each.
// vT layout: [b][h][dh][n].
__global__ __launch_bounds__(256) void k_attn(
    const __bf16* __restrict__ q, const __bf16* __restrict__ k,
    const __bf16* __restrict__ vT, const int* __restrict__ adj,
    __bf16* __restrict__ out)
{
  __shared__ __align__(64) __bf16 sK[2][32][144];   // [key][dh], pitch 144 (pad 8 DW / 64 DW)
  __shared__ __align__(64) __bf16 sVt[2][128][48];  // [dh][key], pitch 48 (pad 8 DW / 16 DW)
  __shared__ __align__(64) __bf16 sP[8][16][40];    // per-wave P staging (16x32)

  const int tid  = threadIdx.x;
  const int lane = tid & 31, wave = tid >> 5;
  const int half = lane >> 4, l16 = lane & 15;
  const int bh = blockIdx.x >> 4;
  const int qg = blockIdx.x & 15;
  const int b  = bh >> 2, h = bh & 3;
  const int q0 = (qg * 8 + wave) * 16;
  const bool issuer = (wave == 0);

  const __bf16* kbase = k  + ((size_t)b * N_ * NH_ + h) * DH_;           // + key*512
  const __bf16* vbase = vT + ((size_t)(b * NH_ + h) * DH_) * N_;         // + dh*N + key

  // preload Q fragments (4 x 16x32 over DH=128); row = q0 + l16
  v16bf aQ[4];
  {
    const __bf16* qrow = q + ((size_t)(b * N_ + q0 + l16) * NH_ + h) * DH_;
#pragma unroll
    for (int c = 0; c < 4; ++c) {
      v8bf lo = *(const v8bf*)(qrow + c * 32 + half * 16);
      v8bf hi = *(const v8bf*)(qrow + c * 32 + half * 16 + 8);
      aQ[c] = concat8(lo, hi);
    }
  }

  v8f o[8];
#pragma unroll
  for (int t = 0; t < 8; ++t) o[t] = (v8f){};
  float m[8], lsum[8];
#pragma unroll
  for (int r = 0; r < 8; ++r) { m[r] = -1e30f; lsum[r] = 0.f; }

  if (issuer) {
    tdm_load_2d(&sK[0][0][0],  kbase, 128, 32, (unsigned long long)(NH_ * DH_), 5, 7);
    tdm_load_2d(&sVt[0][0][0], vbase, 32, 128, (unsigned long long)N_, 3, 7);
    __builtin_amdgcn_s_wait_tensorcnt(0);
  }
  __syncthreads();

  const int nck = N_ / 32;
  for (int ck = 0; ck < nck; ++ck) {
    const int cur = ck & 1;
    if (issuer && (ck + 1) < nck) {
      const int nxt = cur ^ 1;
      tdm_load_2d(&sK[nxt][0][0],  kbase + (size_t)(ck + 1) * 32 * NH_ * DH_,
                  128, 32, (unsigned long long)(NH_ * DH_), 5, 7);
      tdm_load_2d(&sVt[nxt][0][0], vbase + (size_t)(ck + 1) * 32,
                  32, 128, (unsigned long long)N_, 3, 7);
    }

    // scores: S(16x32) = Q(16x128) @ K^T(128x32), two 16x16 tiles
    v8f s0 = {}, s1 = {};
#pragma unroll
    for (int c = 0; c < 4; ++c) {
      v16bf bk0 = *(const v16bf*)&sK[cur][l16][c * 32 + half * 16];
      v16bf bk1 = *(const v16bf*)&sK[cur][16 + l16][c * 32 + half * 16];
      s0 = wmma_bf16(aQ[c], bk0, s0);
      s1 = wmma_bf16(aQ[c], bk1, s1);
    }

    // mask + online softmax (rows live across 16 lanes of a half-wave)
#pragma unroll
    for (int r = 0; r < 8; ++r) {
      int qr = q0 + r + 8 * half;
      const int* arow = adj + ((size_t)b * N_ + qr) * N_ + ck * 32;
      float sv0 = (arow[l16]      == 0) ? -1e9f : s0[r];
      float sv1 = (arow[16 + l16] == 0) ? -1e9f : s1[r];
      float mx = fmaxf(sv0, sv1);
#pragma unroll
      for (int off = 8; off >= 1; off >>= 1)
        mx = fmaxf(mx, __shfl_xor(mx, off, 16));
      float nm = fmaxf(m[r], mx);
      float alpha = __expf(m[r] - nm);
      m[r] = nm;
      float p0 = __expf(sv0 - nm);
      float p1 = __expf(sv1 - nm);
      lsum[r] = lsum[r] * alpha + p0 + p1;
#pragma unroll
      for (int t = 0; t < 8; ++t) o[t][r] *= alpha;
      sP[wave][r + 8 * half][l16]      = (__bf16)p0;
      sP[wave][r + 8 * half][16 + l16] = (__bf16)p1;
    }
    asm volatile("" ::: "memory");     // LDS in-order within wave; block compiler reordering

    // O += P(16x32) @ V(32x128)
    v8bf plo = *(const v8bf*)&sP[wave][l16][half * 16];
    v8bf phi = *(const v8bf*)&sP[wave][l16][half * 16 + 8];
    v16bf aP = concat8(plo, phi);
#pragma unroll
    for (int t = 0; t < 8; ++t) {
      v16bf bv = *(const v16bf*)&sVt[cur][t * 16 + l16][half * 16];
      o[t] = wmma_bf16(aP, bv, o[t]);
    }

    if (issuer && (ck + 1) < nck) __builtin_amdgcn_s_wait_tensorcnt(0);
    __syncthreads();
  }

  // finalize: divide by row sums, store bf16
#pragma unroll
  for (int r = 0; r < 8; ++r) {
    float d = lsum[r];
#pragma unroll
    for (int off = 8; off >= 1; off >>= 1)
      d += __shfl_xor(d, off, 16);
    float inv = 1.f / d;
    int row = q0 + r + 8 * half;
    __bf16* dst = out + ((size_t)(b * N_ + row) * NH_ + h) * DH_;
#pragma unroll
    for (int t = 0; t < 8; ++t)
      dst[t * 16 + l16] = (__bf16)(o[t][r] * inv);
  }
}

// ---------------- LayerNorm (optional residual), one wave per row ----------------
__global__ __launch_bounds__(256) void k_ln(
    const float* __restrict__ src, const float* __restrict__ res,
    const float* __restrict__ g, const float* __restrict__ bta,
    float* __restrict__ outF, __bf16* __restrict__ outB)
{
  const int lane = threadIdx.x & 31, wave = threadIdx.x >> 5;
  const int row = blockIdx.x * 8 + wave;
  const float* sr = src + (size_t)row * H_;
  const float* rr = res ? res + (size_t)row * H_ : nullptr;
  float vals[16];
  float s = 0.f, s2 = 0.f;
#pragma unroll
  for (int i = 0; i < 16; ++i) {
    float x = sr[i * 32 + lane];
    if (rr) x += rr[i * 32 + lane];
    vals[i] = x; s += x; s2 += x * x;
  }
#pragma unroll
  for (int off = 16; off >= 1; off >>= 1) {
    s  += __shfl_xor(s,  off, 32);
    s2 += __shfl_xor(s2, off, 32);
  }
  float mu   = s * (1.f / H_);
  float var  = s2 * (1.f / H_) - mu * mu;
  float rstd = rsqrtf(var + 1e-5f);
#pragma unroll
  for (int i = 0; i < 16; ++i) {
    int col = i * 32 + lane;
    float y = (vals[i] - mu) * rstd * g[col] + bta[col];
    outF[(size_t)row * H_ + col] = y;
    if (outB) outB[(size_t)row * H_ + col] = (__bf16)y;
  }
}

// ---------------- final packing: glob | gents | emask ----------------
__global__ void k_final(const float* __restrict__ xF, float* __restrict__ out, int total) {
  int i = blockIdx.x * 256 + threadIdx.x;
  if (i >= total) return;
  const int nglob = B_ * H_;
  const int ngent = B_ * N_ * H_;
  if (i < nglob) {
    int b = i / H_, c = i % H_;
    out[i] = xF[((size_t)b * N_ + E_) * H_ + c];
  } else if (i < nglob + ngent) {
    out[i] = xF[i - nglob];
  } else {
    out[i] = 1.0f;   // emask: arange(N) <= N -> all true
  }
}

extern "C" void kernel_launch(void* const* d_in, const int* in_sizes, int n_in,
                              void* d_out, int out_size, void* d_ws, size_t ws_size,
                              hipStream_t stream) {
  (void)in_sizes; (void)n_in; (void)out_size; (void)ws_size;
  const int*   adjs  = (const int*)d_in[0];
  const int*   rels  = (const int*)d_in[1];
  const float* vents = (const float*)d_in[2];
  const float* renc  = (const float*)d_in[4];
  const float* Wq = (const float*)d_in[5];
  const float* Wk = (const float*)d_in[6];
  const float* Wv = (const float*)d_in[7];
  const float* Wo = (const float*)d_in[8];
  const float* W1 = (const float*)d_in[9];
  const float* W2 = (const float*)d_in[10];
  const float* bq = (const float*)d_in[11];
  const float* bk = (const float*)d_in[12];
  const float* bv = (const float*)d_in[13];
  const float* bo = (const float*)d_in[14];
  const float* b1 = (const float*)d_in[15];
  const float* b2 = (const float*)d_in[16];
  const float* ln1_g = (const float*)d_in[17];
  const float* ln1_b = (const float*)d_in[18];
  const float* ln2_g = (const float*)d_in[19];
  const float* ln2_b = (const float*)d_in[20];
  const float* prelu = (const float*)d_in[21];

  char* ws = (char*)d_ws;
  size_t off = 0;
  auto alloc = [&](size_t bytes) -> char* {
    char* p = ws + off;
    off = (off + bytes + 255) & ~(size_t)255;
    return p;
  };
  const size_t nMH  = (size_t)M_ * H_;
  const size_t nMH4 = (size_t)M_ * H4_;
  float*  xF    = (float*)alloc(nMH * 4);
  __bf16* xB    = (__bf16*)alloc(nMH * 2);
  __bf16* qB    = (__bf16*)alloc(nMH * 2);
  __bf16* kB    = (__bf16*)alloc(nMH * 2);
  __bf16* vTB   = (__bf16*)alloc(nMH * 2);
  __bf16* oB    = (__bf16*)alloc(nMH * 2);
  float*  attnF = (float*)alloc(nMH * 4);
  float*  tF    = (float*)alloc(nMH * 4);
  __bf16* tB    = (__bf16*)alloc(nMH * 2);
  __bf16* hB    = (__bf16*)alloc(nMH4 * 2);
  float*  yF    = (float*)alloc(nMH * 4);
  const size_t nW  = (size_t)P_ * H_ * H_;
  const size_t nW1 = (size_t)P_ * H_ * H4_;
  __bf16* WqT = (__bf16*)alloc(nW * 2);
  __bf16* WkT = (__bf16*)alloc(nW * 2);
  __bf16* WvT = (__bf16*)alloc(nW * 2);
  __bf16* WoT = (__bf16*)alloc(nW * 2);
  __bf16* W1T = (__bf16*)alloc(nW1 * 2);
  __bf16* W2T = (__bf16*)alloc(nW1 * 2);

  // cast + transpose weights (bf16, [p][n][k])
  dim3 gHH((H_ * H_ + 255) / 256, P_);
  dim3 gHH4((H_ * H4_ + 255) / 256, P_);
  k_cast_t<<<gHH,  256, 0, stream>>>(Wq, WqT, H_, H_);
  k_cast_t<<<gHH,  256, 0, stream>>>(Wk, WkT, H_, H_);
  k_cast_t<<<gHH,  256, 0, stream>>>(Wv, WvT, H_, H_);
  k_cast_t<<<gHH,  256, 0, stream>>>(Wo, WoT, H_, H_);
  k_cast_t<<<gHH4, 256, 0, stream>>>(W1, W1T, H_, H4_);
  k_cast_t<<<gHH4, 256, 0, stream>>>(W2, W2T, H4_, H_);

  // x0
  k_build_x<<<(B_ * N_ * H_ + 255) / 256, 256, 0, stream>>>(vents, renc, rels, xF, xB);

  const float qscale = 0.08838834764831845f;  // 1/sqrt(DH)
  for (int j = 0; j < P_; ++j) {
    const size_t wo  = (size_t)j * H_ * H_;
    const size_t w1o = (size_t)j * H_ * H4_;
    dim3 gH(H_ / 64, M_ / 64);    // 8 x 128
    dim3 gH4(H4_ / 64, M_ / 64);  // 32 x 128

    k_gemm<<<gH, 256, 0, stream>>>(xB, WqT + wo, bq + j * H_, nullptr,
                                   nullptr, qB, M_, H_, H_, qscale, 0);
    k_gemm<<<gH, 256, 0, stream>>>(xB, WkT + wo, bk + j * H_, nullptr,
                                   nullptr, kB, M_, H_, H_, 1.0f, 0);
    k_gemm<<<gH, 256, 0, stream>>>(xB, WvT + wo, bv + j * H_, nullptr,
                                   nullptr, vTB, M_, H_, H_, 1.0f, 1);

    k_attn<<<256, 256, 0, stream>>>(qB, kB, vTB, adjs, oB);

    k_gemm<<<gH, 256, 0, stream>>>(oB, WoT + wo, bo + j * H_, nullptr,
                                   attnF, nullptr, M_, H_, H_, 1.0f, 0);

    k_ln<<<M_ / 8, 256, 0, stream>>>(attnF, nullptr, ln1_g + j * H_, ln1_b + j * H_, tF, tB);

    k_gemm<<<gH4, 256, 0, stream>>>(tB, W1T + w1o, b1 + j * H4_, prelu + j * H4_,
                                    nullptr, hB, M_, H4_, H_, 1.0f, 0);
    k_gemm<<<gH, 256, 0, stream>>>(hB, W2T + w1o, b2 + j * H_, nullptr,
                                   yF, nullptr, M_, H_, H4_, 1.0f, 0);

    k_ln<<<M_ / 8, 256, 0, stream>>>(yF, tF, ln2_g + j * H_, ln2_b + j * H_, xF, xB);
  }

  const int total = B_ * H_ + B_ * N_ * H_ + B_ * N_;
  k_final<<<(total + 255) / 256, 256, 0, stream>>>(xF, (float*)d_out, total);
}
